// SKCLite_65025804861748
// MI455X (gfx1250) — compile-verified, matrix operand order checked by reading, not run on hardware
//
#include <hip/hip_runtime.h>
#include <hip/hip_bf16.h>

// ---------------------------------------------------------------------------
// SKCLite cross-attention for MI455X (gfx1250, wave32, WMMA).
//
// B=4, C=256, N=H*W=4096, R=8 (rank), G=4 groups (CG=64).
// Dominant cost: out = v @ softmax(q^T k)  -> 2 * 2*B*C*N^2 = 68.7 GFLOP.
// Strategy: flash-style on-the-fly score tiles.
//   scores (rank-8) generated with V_WMMA_F32_16X16X4_F32,
//   exp() + bf16-pack lands *exactly* in the B-operand layout of
//   V_WMMA_F32_16X16X32_BF16 (f32 C/D layout == bf16 B layout modulo packing),
//   main GEMM in bf16 WMMA with f32 accumulation.
// ---------------------------------------------------------------------------

#define NB   4      // batch
#define NC   256    // channels
#define NN   4096   // H*W
#define NR   8      // rank
#define NG   4      // groups
#define NCG  64     // channels per group
#define LOG2E 1.4426950408889634f

typedef __attribute__((ext_vector_type(2)))  float        v2f;
typedef __attribute__((ext_vector_type(8)))  float        v8f;
typedef __attribute__((ext_vector_type(16))) __bf16       v16bf;
typedef __attribute__((ext_vector_type(4)))  unsigned int v4u;

// ---------------------------------------------------------------------------
// Kernel 1: q/k projections.  q = (qw . f + qb) * log2(e)   (exp -> exp2)
//           k =  kw . f + kb
// Layout: qbuf/kbuf = [dir][B][R][N] f32.
// dir0 (LU path): q from fL (qL_w), k from fU (kU_w)
// dir1 (UL path): q from fU (qU_w), k from fL (kL_w)
// ---------------------------------------------------------------------------
__global__ __launch_bounds__(256)
void SKC_proj(const float* __restrict__ fL, const float* __restrict__ fU,
              const float* __restrict__ qLw, const float* __restrict__ qLb,
              const float* __restrict__ kUw, const float* __restrict__ kUb,
              const float* __restrict__ qUw, const float* __restrict__ qUb,
              const float* __restrict__ kLw, const float* __restrict__ kLb,
              float* __restrict__ qbuf, float* __restrict__ kbuf)
{
    const int t = blockIdx.x * 256 + threadIdx.x;   // over B*N
    const int b = t / NN;
    const int n = t % NN;

    float aq0[NR], ak0[NR], aq1[NR], ak1[NR];
#pragma unroll
    for (int r = 0; r < NR; ++r) { aq0[r] = 0.f; ak0[r] = 0.f; aq1[r] = 0.f; ak1[r] = 0.f; }

    const float* fl = fL + (size_t)b * NC * NN + n;
    const float* fu = fU + (size_t)b * NC * NN + n;
    for (int c = 0; c < NC; ++c) {
        const float xl = fl[(size_t)c * NN];
        const float xu = fu[(size_t)c * NN];
#pragma unroll
        for (int r = 0; r < NR; ++r) {
            aq0[r] += qLw[r * NC + c] * xl;   // uniform -> scalar loads
            ak0[r] += kUw[r * NC + c] * xu;
            aq1[r] += qUw[r * NC + c] * xu;
            ak1[r] += kLw[r * NC + c] * xl;
        }
    }
#pragma unroll
    for (int r = 0; r < NR; ++r) {
        qbuf[((size_t)0 * NB + b) * (NR * NN) + r * NN + n] = (aq0[r] + qLb[r]) * LOG2E;
        kbuf[((size_t)0 * NB + b) * (NR * NN) + r * NN + n] =  ak0[r] + kUb[r];
        qbuf[((size_t)1 * NB + b) * (NR * NN) + r * NN + n] = (aq1[r] + qUb[r]) * LOG2E;
        kbuf[((size_t)1 * NB + b) * (NR * NN) + r * NN + n] =  ak1[r] + kLb[r];
    }
}

// ---------------------------------------------------------------------------
// Kernel 2: softmax normalizers.  invZ[dir][b][m] = 1 / sum_j exp(q_m . k_j)
// One wave per 16-row m-tile.  A operand (q, fixed m) stays in registers;
// loop j in tiles of 16: two chained V_WMMA_F32_16X16X4_F32 (K=8), exp2,
// accumulate per-VGPR, then 16-lane xor reduction (lane halves independent).
// ---------------------------------------------------------------------------
__global__ __launch_bounds__(256)
void SKC_stats(const float* __restrict__ qbuf, const float* __restrict__ kbuf,
               float* __restrict__ invZ)
{
    const int lane = threadIdx.x & 31;
    const int wave = threadIdx.x >> 5;
    const int b    = blockIdx.y;
    const int dir  = blockIdx.z;
    const int m0   = (blockIdx.x * 8 + wave) * 16;
    const int l15  = lane & 15;
    const int hi   = lane >> 4;                 // lane-half select

    const float* q = qbuf + ((size_t)dir * NB + b) * (NR * NN);
    const float* k = kbuf + ((size_t)dir * NB + b) * (NR * NN);

    // A fragment: 16x4 f32: lanes 0-15 hold K=0,1 ; lanes 16-31 hold K=2,3
    v2f aq0, aq1;
    {
        const int m = m0 + l15;
        const int r = hi * 2;
        aq0.x = q[(r + 0) * NN + m];  aq0.y = q[(r + 1) * NN + m];
        aq1.x = q[(r + 4) * NN + m];  aq1.y = q[(r + 5) * NN + m];
    }

    v8f zacc = {0.f, 0.f, 0.f, 0.f, 0.f, 0.f, 0.f, 0.f};
    for (int j0 = 0; j0 < NN; j0 += 16) {
        v2f bk0, bk1;
        const int n = j0 + l15;
        const int r = hi * 2;
        bk0.x = k[(r + 0) * NN + n];  bk0.y = k[(r + 1) * NN + n];
        bk1.x = k[(r + 4) * NN + n];  bk1.y = k[(r + 5) * NN + n];

        v8f s = {0.f, 0.f, 0.f, 0.f, 0.f, 0.f, 0.f, 0.f};
        s = __builtin_amdgcn_wmma_f32_16x16x4_f32(false, aq0, false, bk0, (short)0, s, false, false);
        s = __builtin_amdgcn_wmma_f32_16x16x4_f32(false, aq1, false, bk1, (short)0, s, false, false);
#pragma unroll
        for (int g = 0; g < 8; ++g) zacc[g] += __builtin_amdgcn_exp2f(s[g]);
    }
    // reduce across the 16 lanes of each half (j is the lane dimension)
#pragma unroll
    for (int g = 0; g < 8; ++g) {
        float z = zacc[g];
        z += __shfl_xor(z, 1, 32);
        z += __shfl_xor(z, 2, 32);
        z += __shfl_xor(z, 4, 32);
        z += __shfl_xor(z, 8, 32);
        if (l15 == 0) {
            const int m = m0 + g + hi * 8;      // D layout: VGPR g -> row g / g+8
            invZ[((size_t)dir * NB + b) * NN + m] = 1.0f / z;
        }
    }
}

// ---------------------------------------------------------------------------
// Kernel 3: grouped v-projection fused with 1/Z, emitted as bf16.
// vp[dir][b][c][m] = bf16( (vb[c] + sum_i vw[c][i] f[b,g*64+i,m]) * invZ[dir][b][m] )
// Thread = (dir,b,g,n); the 64-deep f column lives in registers, read once.
// ---------------------------------------------------------------------------
__global__ __launch_bounds__(256)
void SKC_vp(const float* __restrict__ fL, const float* __restrict__ fU,
            const float* __restrict__ vUw, const float* __restrict__ vUb,
            const float* __restrict__ vLw, const float* __restrict__ vLb,
            const float* __restrict__ invZ, __bf16* __restrict__ vp)
{
    const int dir = blockIdx.z;
    const int bg  = blockIdx.y;
    const int b   = bg >> 2;
    const int g   = bg & 3;
    const int n   = blockIdx.x * 256 + threadIdx.x;

    const float* f  = (dir == 0) ? fU  : fL;    // LU uses vU(fU); UL uses vL(fL)
    const float* vw = (dir == 0) ? vUw : vLw;
    const float* vb = (dir == 0) ? vUb : vLb;

    float fr[NCG];
#pragma unroll
    for (int i = 0; i < NCG; ++i)
        fr[i] = f[((size_t)b * NC + g * NCG + i) * NN + n];

    const float iz = invZ[((size_t)dir * NB + b) * NN + n];

    for (int co = 0; co < NCG; ++co) {
        const int c = g * NCG + co;
        float acc = vb[c];
#pragma unroll
        for (int i = 0; i < NCG; ++i)
            acc += vw[(size_t)c * NCG + i] * fr[i];   // uniform -> scalar loads
        vp[(((size_t)dir * NB + b) * NC + c) * NN + n] = (__bf16)(acc * iz);
    }
}

// ---------------------------------------------------------------------------
// Kernel 4: main attend + residual.
//   out[c,n] = f[c,n] + beta * sum_m vp[c,m] * exp(q_m . k_n)
// Wave owns a fixed 16-wide n-tile and all 16 c-tiles (C=256).
// Per 32-wide m-step: 4x f32 WMMA score tiles -> exp2 -> bf16 pack
// (f32 D layout coincides with bf16 B layout: zero lane shuffles) ->
// 16x V_WMMA_F32_16X16X32_BF16 with two b128 vp loads each.
// ---------------------------------------------------------------------------
__global__ __launch_bounds__(256)
void SKC_attend(const float* __restrict__ fL, const float* __restrict__ fU,
                const float* __restrict__ qbuf, const float* __restrict__ kbuf,
                const __bf16* __restrict__ vp, const float* __restrict__ betap,
                float* __restrict__ out)
{
    const int lane = threadIdx.x & 31;
    const int wave = threadIdx.x >> 5;
    const int b    = blockIdx.y;
    const int dir  = blockIdx.z;
    const int n0   = (blockIdx.x * 8 + wave) * 16;
    const int l15  = lane & 15;
    const int hi   = lane >> 4;
    const float beta = *betap;

    const float*  q   = qbuf + ((size_t)dir * NB + b) * (NR * NN);
    const float*  k   = kbuf + ((size_t)dir * NB + b) * (NR * NN);
    const __bf16* vpb = vp   + ((size_t)dir * NB + b) * (size_t)(NC * NN);

    // k B-fragment for this n-tile: constant across the whole m loop.
    v2f bk0, bk1;
    {
        const int n = n0 + l15;
        const int r = hi * 2;
        bk0.x = k[(r + 0) * NN + n];  bk0.y = k[(r + 1) * NN + n];
        bk1.x = k[(r + 4) * NN + n];  bk1.y = k[(r + 5) * NN + n];
    }

    v8f acc[16];
    const v8f vzero = {0.f, 0.f, 0.f, 0.f, 0.f, 0.f, 0.f, 0.f};
#pragma unroll
    for (int ct = 0; ct < 16; ++ct) acc[ct] = vzero;

    const int moff = hi * 8;   // bf16 A/B layouts: lane-half K offset

    for (int m0 = 0; m0 < NN; m0 += 32) {
        // ---- score tiles s0 (m0..m0+15) and s1 (m0+16..m0+31), K=8 rank ----
        v2f a0lo, a0hi, a1lo, a1hi;
        {
            const int r  = hi * 2;
            const int mA = m0 + l15;
            const int mB = m0 + 16 + l15;
            a0lo.x = q[(r + 0) * NN + mA];  a0lo.y = q[(r + 1) * NN + mA];
            a0hi.x = q[(r + 4) * NN + mA];  a0hi.y = q[(r + 5) * NN + mA];
            a1lo.x = q[(r + 0) * NN + mB];  a1lo.y = q[(r + 1) * NN + mB];
            a1hi.x = q[(r + 4) * NN + mB];  a1hi.y = q[(r + 5) * NN + mB];
        }
        v8f s0 = vzero, s1 = vzero;
        s0 = __builtin_amdgcn_wmma_f32_16x16x4_f32(false, a0lo, false, bk0, (short)0, s0, false, false);
        s0 = __builtin_amdgcn_wmma_f32_16x16x4_f32(false, a0hi, false, bk1, (short)0, s0, false, false);
        s1 = __builtin_amdgcn_wmma_f32_16x16x4_f32(false, a1lo, false, bk0, (short)0, s1, false, false);
        s1 = __builtin_amdgcn_wmma_f32_16x16x4_f32(false, a1hi, false, bk1, (short)0, s1, false, false);

        // ---- E tile -> bf16 B operand (layouts line up; in-lane pack) ----
        v16bf eb;
#pragma unroll
        for (int g = 0; g < 8; ++g) {
            eb[g]     = (__bf16)__builtin_amdgcn_exp2f(s0[g]);   // K = moff+g
            eb[8 + g] = (__bf16)__builtin_amdgcn_exp2f(s1[g]);   // K = 16+moff+g
        }

        // ---- 16 c-tiles of the main bf16 GEMM ----
        const __bf16* vpl = vpb + (size_t)l15 * NN + m0 + moff;
#pragma unroll
        for (int ct = 0; ct < 16; ++ct) {
            const __bf16* p = vpl + (size_t)ct * 16 * NN;
            union { v4u u[2]; v16bf h; } U;
            U.u[0] = *(const v4u*)(p);        // K = moff .. moff+7     (16B aligned)
            U.u[1] = *(const v4u*)(p + 16);   // K = 16+moff .. 16+moff+7
            acc[ct] = __builtin_amdgcn_wmma_f32_16x16x32_bf16(
                false, U.h, false, eb, (short)0, acc[ct], false, false);
        }
    }

    // ---- epilogue: fused residual, write d_out ----
    const float* f = (dir == 0) ? fL : fU;
    float* o = out + (size_t)dir * ((size_t)NB * NC * NN);
#pragma unroll
    for (int ct = 0; ct < 16; ++ct) {
#pragma unroll
        for (int g = 0; g < 8; ++g) {
            const int c = ct * 16 + g + hi * 8;
            const int n = n0 + l15;
            const size_t idx = ((size_t)b * NC + c) * NN + n;
            o[idx] = f[idx] + beta * acc[ct][g];
        }
    }
}

// ---------------------------------------------------------------------------
// Host-side launcher.  Workspace layout (floats unless noted):
//   qbuf : 2*B*R*N            = 262144
//   kbuf : 2*B*R*N            = 262144
//   invZ : 2*B*N              =  32768
//   vp   : 2*B*C*N  (bf16)    -> 16.8 MB      (total ~19 MB)
// ---------------------------------------------------------------------------
extern "C" void kernel_launch(void* const* d_in, const int* in_sizes, int n_in,
                              void* d_out, int out_size, void* d_ws, size_t ws_size,
                              hipStream_t stream)
{
    const float* fL   = (const float*)d_in[0];
    const float* fU   = (const float*)d_in[1];
    const float* qLw  = (const float*)d_in[2];
    const float* qLb  = (const float*)d_in[3];
    const float* kUw  = (const float*)d_in[4];
    const float* kUb  = (const float*)d_in[5];
    const float* vUw  = (const float*)d_in[6];
    const float* vUb  = (const float*)d_in[7];
    const float* qUw  = (const float*)d_in[8];
    const float* qUb  = (const float*)d_in[9];
    const float* kLw  = (const float*)d_in[10];
    const float* kLb  = (const float*)d_in[11];
    const float* vLw  = (const float*)d_in[12];
    const float* vLb  = (const float*)d_in[13];
    const float* beta = (const float*)d_in[14];
    float* outp = (float*)d_out;

    float*  qbuf = (float*)d_ws;
    float*  kbuf = qbuf + (size_t)2 * NB * NR * NN;
    float*  invZ = kbuf + (size_t)2 * NB * NR * NN;
    __bf16* vp   = (__bf16*)(invZ + (size_t)2 * NB * NN);

    // 1) projections (q pre-scaled by log2 e)
    SKC_proj<<<dim3((NB * NN) / 256), dim3(256), 0, stream>>>(
        fL, fU, qLw, qLb, kUw, kUb, qUw, qUb, kLw, kLb, qbuf, kbuf);

    // 2) softmax normalizers via f32 WMMA score tiles
    SKC_stats<<<dim3(NN / 128, NB, 2), dim3(256), 0, stream>>>(qbuf, kbuf, invZ);

    // 3) grouped v-projection folded with 1/Z, bf16 output
    SKC_vp<<<dim3(NN / 256, NB * NG, 2), dim3(256), 0, stream>>>(
        fL, fU, vUw, vUb, vLw, vLb, invZ, vp);

    // 4) main attend (f32-WMMA score gen + bf16-WMMA GEMM) + fused residual
    SKC_attend<<<dim3(NN / 128, NB, 2), dim3(256), 0, stream>>>(
        fL, fU, qbuf, kbuf, vp, beta, outp);
}
